// MoEElementFusion_28630251995586
// MI455X (gfx1250) — matrix operand order repaired and checked by reading, not compiled, not run on hardware
//
#include <hip/hip_runtime.h>
#include <hip/hip_bf16.h>
#include <math.h>

// ---------------------------------------------------------------------------
// Problem constants (from the reference)
// ---------------------------------------------------------------------------
#define D_MODEL 256
#define D_FF    1024
#define N_EXP   16
#define N_VIEWS 3
#define N_TOK   8192
#define TOPK    4
#define TILE_M  16
#define WPB     4      // waves per block (share weight streams via WGP$)

// LDS row strides (elements); rows stay 16-byte aligned
#define SV_STRIDE  264   // 256 + 8 pad;  528B  = 33*16
#define SHC_STRIDE 40    // 32  + 8 pad;  80B   = 5*16

typedef __attribute__((ext_vector_type(16))) __bf16 v16bf;
typedef __attribute__((ext_vector_type(8)))  __bf16 v8bf;
typedef __attribute__((ext_vector_type(8)))  float  v8f;

typedef __attribute__((ext_vector_type(4))) unsigned int u32x4;
typedef __attribute__((ext_vector_type(8))) int          i32x8;
typedef __attribute__((ext_vector_type(4))) int          i32x4;

union FragU { v16bf w; v8bf h[2]; };

// A-matrix 16x32 bf16 fragment (ISA 7.12.2): half 0 -> K {k0..k0+7, k0+16..k0+23}
//                                            half 1 -> K {k0+8..k0+15, k0+24..k0+31}
__device__ __forceinline__ v16bf load_a_frag(const __bf16* row, int kbase, int half) {
  FragU u;
  const __bf16* p = row + kbase + half * 8;
  u.h[0] = *(const v8bf*)(p);
  u.h[1] = *(const v8bf*)(p + 16);
  return u.w;
}

// B-matrix 32x16 bf16 fragment: lane = column N; same K order as A:
// half 0 -> K {k0..k0+7, k0+16..k0+23}, half 1 -> K {k0+8..k0+15, k0+24..k0+31}.
__device__ __forceinline__ v16bf load_b_frag(const __bf16* row, int kbase, int half) {
  FragU u;
  const __bf16* p = row + kbase + half * 8;
  u.h[0] = *(const v8bf*)(p);
  u.h[1] = *(const v8bf*)(p + 16);
  return u.w;
}

// ---------------------------------------------------------------------------
// Tensor Data Mover: DMA a 16x256 bf16 tile (row stride 256) from global into
// LDS with 4-DWORD padding after every 128-DWORD row -> 264-element LDS stride.
// D# layout per CDNA5 ISA.
// ---------------------------------------------------------------------------
#if __has_include(<hip/amd_detail/amd_gfx1250_TDM.h>)
#define TDM_6ARG 1
#else
#define TDM_6ARG 0
#endif

__device__ __forceinline__ void tdm_load_tile_16x256(const __bf16* gsrc, __bf16* lds_dst) {
  unsigned lds_addr = (unsigned)(size_t)(void*)lds_dst;  // low 32 bits = LDS offset
  unsigned long long ga = (unsigned long long)(size_t)(const void*)gsrc;
  u32x4 g0;
  g0[0] = 1u;                                           // count=1 (valid), user mode
  g0[1] = lds_addr;                                     // lds_addr
  g0[2] = (unsigned)(ga & 0xffffffffu);                 // global_addr[31:0]
  g0[3] = (unsigned)((ga >> 32) & 0x1ffffffu) | (2u << 30);  // global_addr[56:32], type=2
  i32x8 g1;
  // data_size=1(2B), pad_enable=1, pad_interval=6(128 DW), pad_amount=3(4 DW)
  g1[0] = (int)0x07910000;
  g1[1] = (int)(256u << 16);   // tensor_dim0[15:0]=256 in bits[31:16]
  g1[2] = (int)(16u  << 16);   // tensor_dim0 hi=0 | tensor_dim1[15:0]=16
  g1[3] = (int)(256u << 16);   // tensor_dim1 hi=0 | tile_dim0=256
  g1[4] = 16;                  // tile_dim1=16, tile_dim2=0
  g1[5] = 256;                 // tensor_dim0_stride[31:0]=256
  g1[6] = 0;                   // stride hi / tensor_dim1_stride lo
  g1[7] = 0;
  i32x4 z4 = {0, 0, 0, 0};
#if TDM_6ARG
  i32x8 z8 = {0, 0, 0, 0, 0, 0, 0, 0};
  __builtin_amdgcn_tensor_load_to_lds(g0, g1, z4, z4, z8, 0);
#else
  __builtin_amdgcn_tensor_load_to_lds(g0, g1, z4, z4, 0);
#endif
}

// ---------------------------------------------------------------------------
// Kernel 1: fp32 views -> bf16 workspace  [3][N_TOK][D_MODEL]
// ---------------------------------------------------------------------------
__global__ void cvt_views_kernel(const float* __restrict__ v0,
                                 const float* __restrict__ v1,
                                 const float* __restrict__ v2,
                                 __bf16* __restrict__ vbf) {
  size_t i = (size_t)blockIdx.x * blockDim.x + threadIdx.x;
  const size_t n = (size_t)N_TOK * D_MODEL;
  if (i >= (size_t)N_VIEWS * n) return;
  int view = (int)(i / n);
  size_t off = i % n;
  const float* src = (view == 0) ? v0 : (view == 1 ? v1 : v2);
  vbf[i] = (__bf16)src[off];
}

// Kernel 2a: W1 [E][256][1024] fp32 -> W1^T bf16 [E][1024][256]
__global__ void cvt_w1_kernel(const float* __restrict__ W1, __bf16* __restrict__ w1t) {
  size_t i = (size_t)blockIdx.x * blockDim.x + threadIdx.x;
  if (i >= (size_t)N_EXP * D_FF * D_MODEL) return;
  int k = (int)(i % D_MODEL);
  int nn = (int)((i / D_MODEL) % D_FF);
  int e = (int)(i / ((size_t)D_MODEL * D_FF));
  w1t[i] = (__bf16)W1[((size_t)e * D_MODEL + k) * D_FF + nn];
}

// Kernel 2b: W2 [E][1024][256] fp32 -> W2^T bf16 [E][256][1024]
__global__ void cvt_w2_kernel(const float* __restrict__ W2, __bf16* __restrict__ w2t) {
  size_t i = (size_t)blockIdx.x * blockDim.x + threadIdx.x;
  if (i >= (size_t)N_EXP * D_MODEL * D_FF) return;
  int k = (int)(i % D_FF);
  int nn = (int)((i / D_FF) % D_MODEL);
  int e = (int)(i / ((size_t)D_FF * D_MODEL));
  w2t[i] = (__bf16)W2[((size_t)e * D_FF + k) * D_MODEL + nn];
}

// ---------------------------------------------------------------------------
// Kernel 3: Laplace gate, one wave32 per (view, token)
// logits_e = -(|v|^2 + |k_e|^2 - 2 v.k_e) + v.rw_e ; top-4 softmax -> dense[16]
// ---------------------------------------------------------------------------
__global__ __launch_bounds__(128) void gate_kernel(
    const float* __restrict__ v0, const float* __restrict__ v1,
    const float* __restrict__ v2,
    const float* __restrict__ rw,      // [3][16][256]
    const float* __restrict__ ek,      // [16][256]
    float* __restrict__ gates)         // [3][N_TOK][16] dense
{
  const int lane = threadIdx.x & 31;
  const int wid  = blockIdx.x * (blockDim.x >> 5) + (threadIdx.x >> 5);
  if (wid >= N_VIEWS * N_TOK) return;
  const int view = wid / N_TOK;
  const int tok  = wid % N_TOK;
  const float* v = ((view == 0) ? v0 : (view == 1 ? v1 : v2)) + (size_t)tok * D_MODEL;

  float vd[8];
  float vv = 0.f;
#pragma unroll
  for (int j = 0; j < 8; ++j) { vd[j] = v[lane + 32 * j]; vv += vd[j] * vd[j]; }
#pragma unroll
  for (int o = 16; o > 0; o >>= 1) vv += __shfl_xor(vv, o, 32);

  float logits[N_EXP];
#pragma unroll 1
  for (int e = 0; e < N_EXP; ++e) {
    const float* kp = ek + (size_t)e * D_MODEL;
    const float* rp = rw + ((size_t)view * N_EXP + e) * D_MODEL;
    float dvk = 0.f, dvr = 0.f, kk = 0.f;
#pragma unroll
    for (int j = 0; j < 8; ++j) {
      float kv = kp[lane + 32 * j];
      float rv = rp[lane + 32 * j];
      dvk += vd[j] * kv;
      kk  += kv * kv;
      dvr += vd[j] * rv;
    }
#pragma unroll
    for (int o = 16; o > 0; o >>= 1) {
      dvk += __shfl_xor(dvk, o, 32);
      kk  += __shfl_xor(kk,  o, 32);
      dvr += __shfl_xor(dvr, o, 32);
    }
    logits[e] = -(vv + kk - 2.f * dvk) + dvr;
  }

  float tmp[N_EXP];
#pragma unroll
  for (int i = 0; i < N_EXP; ++i) tmp[i] = logits[i];
  float topv[TOPK]; int topi[TOPK];
#pragma unroll
  for (int j = 0; j < TOPK; ++j) {
    float best = tmp[0]; int bi = 0;
#pragma unroll
    for (int i = 1; i < N_EXP; ++i) if (tmp[i] > best) { best = tmp[i]; bi = i; }
    topv[j] = best; topi[j] = bi; tmp[bi] = -3.4e38f;
  }
  float m = topv[0], s = 0.f, ex[TOPK];
#pragma unroll
  for (int j = 0; j < TOPK; ++j) { ex[j] = expf(topv[j] - m); s += ex[j]; }
  float inv = 1.f / s;
  if (lane < N_EXP) {
    float g = 0.f;
#pragma unroll
    for (int j = 0; j < TOPK; ++j) if (topi[j] == lane) g = ex[j] * inv;
    gates[((size_t)view * N_TOK + tok) * N_EXP + lane] = g;
  }
}

// ---------------------------------------------------------------------------
// Kernel 4: fused MoE FFN. 4 waves/block, one 16-token tile per wave; the four
// waves stream identical weight fragments (WGP$ reuse). H is consumed in
// K-chunks of 32 so no big LDS H buffer is needed:
//   per chunk: 2x(16x16) H tiles = 16 WMMA over V, gelu -> LDS chunk,
//              then 16 WMMA accumulating Y into persistent c2 registers.
// acc[16x256 f32 in VGPRs] += gate_e .* (gelu(V W1e + b1e) W2e + b2e)
// ---------------------------------------------------------------------------
__global__ __launch_bounds__(32 * WPB) void moe_kernel(
    const __bf16* __restrict__ vbf,    // [3][N_TOK][D_MODEL]
    const float*  __restrict__ gates,  // [3][N_TOK][N_EXP]
    const __bf16* __restrict__ w1t,    // [E][D_FF][D_MODEL]
    const float*  __restrict__ b1,     // [E][D_FF]
    const __bf16* __restrict__ w2t,    // [E][D_MODEL][D_FF]
    const float*  __restrict__ b2,     // [E][D_MODEL]
    float* __restrict__ out)           // [N_TOK][D_MODEL]
{
  __shared__ __align__(16) __bf16 sV [WPB][TILE_M * SV_STRIDE];   // 4 x 8.25 KB
  __shared__ __align__(16) __bf16 sHc[WPB][TILE_M * SHC_STRIDE];  // 4 x 1.25 KB

  const int lane = threadIdx.x & 31;
  const int wid  = threadIdx.x >> 5;
  const int half = lane >> 4;
  const int l16  = lane & 15;
  const int m0   = (blockIdx.x * WPB + wid) * TILE_M;

  __bf16* mysV = &sV[wid][0];
  __bf16* mysH = &sHc[wid][0];

  v8f acc[D_MODEL / 16];
#pragma unroll
  for (int nt = 0; nt < D_MODEL / 16; ++nt)
#pragma unroll
    for (int r = 0; r < 8; ++r) acc[nt][r] = 0.0f;

#pragma unroll 1
  for (int view = 0; view < N_VIEWS; ++view) {
    // TDM is unordered vs other memory types: drain our ds reads of the
    // previous view's V tile before the DMA overwrites it.
    asm volatile("s_wait_dscnt 0x0" ::: "memory");
    tdm_load_tile_16x256(vbf + ((size_t)view * N_TOK + m0) * D_MODEL, mysV);
    __builtin_amdgcn_s_wait_tensorcnt(0);

#pragma unroll 1
    for (int e = 0; e < N_EXP; ++e) {
      const __bf16* W1e = w1t + (size_t)e * D_FF * D_MODEL;
      const __bf16* W2e = w2t + (size_t)e * D_MODEL * D_FF;
      if (e + 1 < N_EXP) {  // warm next expert's weights (global_prefetch_b8)
        __builtin_prefetch(w1t + (size_t)(e + 1) * D_FF * D_MODEL + lane * 64, 0, 0);
        __builtin_prefetch(w2t + (size_t)(e + 1) * D_MODEL * D_FF + lane * 64, 0, 0);
      }

      // per-token gate for this expert (row M = r + 8*half of this tile)
      float gm[8];
#pragma unroll
      for (int r = 0; r < 8; ++r)
        gm[r] = gates[((size_t)view * N_TOK + m0 + r + 8 * half) * N_EXP + e];

      v8f c2[D_MODEL / 16];  // persistent Y accumulators over K chunks
#pragma unroll
      for (int nt = 0; nt < D_MODEL / 16; ++nt)
#pragma unroll
        for (int r = 0; r < 8; ++r) c2[nt][r] = 0.0f;

#pragma unroll 1
      for (int kt2 = 0; kt2 < D_FF / 32; ++kt2) {
        // ---- H chunk: columns kt2*32 .. kt2*32+31 (two 16x16 tiles) ----
#pragma unroll
        for (int t = 0; t < 2; ++t) {
          const int nt1 = kt2 * 2 + t;
          v8f c;
#pragma unroll
          for (int r = 0; r < 8; ++r) c[r] = 0.0f;
          const __bf16* brow = W1e + (size_t)(nt1 * 16 + l16) * D_MODEL;
#pragma unroll
          for (int kt = 0; kt < D_MODEL / 32; ++kt) {
            v16bf a = load_a_frag(mysV + l16 * SV_STRIDE, kt * 32, half);
            v16bf b = load_b_frag(brow, kt * 32, half);
            c = __builtin_amdgcn_wmma_f32_16x16x32_bf16(false, a, false, b,
                                                        (short)0, c, false, false);
          }
          float b1v = b1[(size_t)e * D_FF + nt1 * 16 + l16];
#pragma unroll
          for (int r = 0; r < 8; ++r) {
            float x = c[r] + b1v;
            float g = 0.5f * x * (1.0f + erff(x * 0.70710678118654752f)); // exact gelu
            mysH[(r + 8 * half) * SHC_STRIDE + t * 16 + l16] = (__bf16)g;
          }
        }
        // ---- Y += Hchunk @ W2e[kchunk, :] ----
        v16bf ah = load_a_frag(mysH + l16 * SHC_STRIDE, 0, half);
#pragma unroll
        for (int nt2 = 0; nt2 < D_MODEL / 16; ++nt2) {
          v16bf b = load_b_frag(W2e + (size_t)(nt2 * 16 + l16) * D_FF, kt2 * 32, half);
          c2[nt2] = __builtin_amdgcn_wmma_f32_16x16x32_bf16(false, ah, false, b,
                                                            (short)0, c2[nt2],
                                                            false, false);
        }
      }

      // ---- epilogue: acc += gate .* (Y + b2) ----
#pragma unroll
      for (int nt2 = 0; nt2 < D_MODEL / 16; ++nt2) {
        float b2v = b2[(size_t)e * D_MODEL + nt2 * 16 + l16];
#pragma unroll
        for (int r = 0; r < 8; ++r) acc[nt2][r] += gm[r] * (c2[nt2][r] + b2v);
      }
    }
  }

  // --- write fused output (fp32), coalesced across lanes ---
#pragma unroll
  for (int nt = 0; nt < D_MODEL / 16; ++nt) {
#pragma unroll
    for (int r = 0; r < 8; ++r) {
      out[(size_t)(m0 + r + 8 * half) * D_MODEL + nt * 16 + l16] = acc[nt][r];
    }
  }
}

// ---------------------------------------------------------------------------
// Launch
// ---------------------------------------------------------------------------
extern "C" void kernel_launch(void* const* d_in, const int* in_sizes, int n_in,
                              void* d_out, int out_size, void* d_ws, size_t ws_size,
                              hipStream_t stream) {
  (void)in_sizes; (void)n_in; (void)out_size;
  const float* v0 = (const float*)d_in[0];
  const float* v1 = (const float*)d_in[1];
  const float* v2 = (const float*)d_in[2];
  const float* rw = (const float*)d_in[3];
  const float* ek = (const float*)d_in[4];
  const float* W1 = (const float*)d_in[5];
  const float* b1 = (const float*)d_in[6];
  const float* W2 = (const float*)d_in[7];
  const float* b2 = (const float*)d_in[8];
  // d_in[9] = top_k (hardwired to 4 to match the reference constants)

  char* ws = (char*)d_ws;
  size_t o = 0;
  __bf16* vbf = (__bf16*)(ws + o); o += (size_t)N_VIEWS * N_TOK * D_MODEL * sizeof(__bf16);
  __bf16* w1t = (__bf16*)(ws + o); o += (size_t)N_EXP * D_FF * D_MODEL * sizeof(__bf16);
  __bf16* w2t = (__bf16*)(ws + o); o += (size_t)N_EXP * D_MODEL * D_FF * sizeof(__bf16);
  float*  gts = (float*)(ws + o);  o += (size_t)N_VIEWS * N_TOK * N_EXP * sizeof(float);
  if (ws_size < o) return;  // workspace too small (deterministic no-op)

  {
    size_t n = (size_t)N_VIEWS * N_TOK * D_MODEL;
    cvt_views_kernel<<<(unsigned)((n + 255) / 256), 256, 0, stream>>>(v0, v1, v2, vbf);
  }
  {
    size_t n = (size_t)N_EXP * D_FF * D_MODEL;
    cvt_w1_kernel<<<(unsigned)((n + 255) / 256), 256, 0, stream>>>(W1, w1t);
    cvt_w2_kernel<<<(unsigned)((n + 255) / 256), 256, 0, stream>>>(W2, w2t);
  }
  {
    int waves = N_VIEWS * N_TOK;  // one wave32 per (view, token)
    gate_kernel<<<(waves + 3) / 4, 128, 0, stream>>>(v0, v1, v2, rw, ek, gts);
  }
  moe_kernel<<<N_TOK / (TILE_M * WPB), 32 * WPB, 0, stream>>>(
      vbf, gts, w1t, b1, w2t, b2, (float*)d_out);
}
